// HiggsAudioDualFFNDecoderLayer_82781199663129
// MI455X (gfx1250) — compile-verified, matrix operand order checked by reading, not run on hardware
//
#include <hip/hip_runtime.h>
#include <hip/hip_bf16.h>
#include <stdint.h>

// ---------------------------------------------------------------------------
// Types / helpers
// ---------------------------------------------------------------------------
typedef __attribute__((ext_vector_type(16))) __bf16 v16bf;
typedef __attribute__((ext_vector_type(8)))  float  v8f;
typedef __attribute__((ext_vector_type(8)))  unsigned short ushort8;
typedef __attribute__((ext_vector_type(4)))  unsigned int   uint4v;
typedef __attribute__((ext_vector_type(8)))  int            int8v;
typedef __attribute__((ext_vector_type(4)))  int            int4v;
typedef unsigned short u16;

__device__ __forceinline__ u16 f2bf(float f) {
    union { float f; uint32_t u; } x; x.f = f;
    uint32_t r = x.u + 0x7FFFu + ((x.u >> 16) & 1u);   // round-to-nearest-even
    return (u16)(r >> 16);
}

union FragU { ushort8 h[2]; v16bf v; };

// CDNA5 16-bit A/B fragment layout: per lane, two contiguous 8-element K
// chunks: [half*8, half*8+8) in VGPR0-3 and [16+half*8, ...) in VGPR4-7.
__device__ __forceinline__ v16bf make_frag(const u16* lo, const u16* hi) {
    FragU u;
    u.h[0] = *(const ushort8*)lo;
    u.h[1] = *(const ushort8*)hi;
    return u.v;
}

__device__ __forceinline__ v8f v8zero() {
    v8f z = {0.f, 0.f, 0.f, 0.f, 0.f, 0.f, 0.f, 0.f};
    return z;
}

// ---------------------------------------------------------------------------
// Tensor Data Mover: async 2-D tile load Global -> LDS (D# per ISA ch.8)
//   tile_d0 x tile_d1 elements of 2 bytes, row stride = ld (elements)
//   gaddr = byte address of tile start, lds_byte = byte offset in LDS
// ---------------------------------------------------------------------------
#if __has_builtin(__builtin_amdgcn_tensor_load_to_lds)
#define HAVE_TDM 1
__device__ __forceinline__ void tdm_load_2d(uint64_t gaddr, uint32_t lds_byte,
                                            int tile_d0, int tile_d1,
                                            int tens_d0, int tens_d1, int ld) {
    uint4v g0;
    g0[0] = 1u;                                     // count=1, user descriptor
    g0[1] = lds_byte;                               // lds_addr
    g0[2] = (uint32_t)gaddr;                        // global_addr[31:0]
    g0[3] = ((uint32_t)(gaddr >> 32) & 0x01FFFFFFu) // global_addr[56:32]
            | (2u << 30);                           // type = 2 ("image")
    int8v g1;
    g1[0] = (int)(1u << 16);                        // wg_mask=0, data_size=1 (2B)
    g1[1] = (int)(((uint32_t)tens_d0 & 0xFFFFu) << 16);            // dim0 lo16
    g1[2] = (int)((((uint32_t)tens_d0 >> 16) & 0xFFFFu)
            | (((uint32_t)tens_d1 & 0xFFFFu) << 16));              // dim0 hi | dim1 lo
    g1[3] = (int)((((uint32_t)tens_d1 >> 16) & 0xFFFFu)
            | (((uint32_t)tile_d0 & 0xFFFFu) << 16));              // dim1 hi | tile_dim0
    g1[4] = (int)((uint32_t)tile_d1 & 0xFFFFu);                    // tile_dim1, tile_dim2=0
    g1[5] = (int)(uint32_t)ld;                                     // dim0_stride[31:0]
    g1[6] = 0;                                                     // stride hi | dim1_stride lo
    g1[7] = 0;
    int4v gz = {0, 0, 0, 0};
#if defined(__clang_major__) && __clang_major__ >= 23
    int8v gz8 = {0, 0, 0, 0, 0, 0, 0, 0};
    __builtin_amdgcn_tensor_load_to_lds(g0, g1, gz, gz, gz8, 0);
#else
    __builtin_amdgcn_tensor_load_to_lds(g0, g1, gz, gz, 0);
#endif
}
#else
#define HAVE_TDM 0
#endif

// ---------------------------------------------------------------------------
// f32 -> bf16 weight conversion (grid-stride)
// ---------------------------------------------------------------------------
__global__ void cvt_f32_bf16_kernel(const float* __restrict__ src,
                                    u16* __restrict__ dst, long n) {
    long i = (long)blockIdx.x * blockDim.x + threadIdx.x;
    long stride = (long)gridDim.x * blockDim.x;
    for (; i < n; i += stride) dst[i] = f2bf(src[i]);
}

// ---------------------------------------------------------------------------
// RMSNorm with per-token dual-weight select (mask ? w1 : w0) -> bf16 out
// ---------------------------------------------------------------------------
__global__ __launch_bounds__(256)
void rmsnorm_dual_kernel(const float* __restrict__ x,
                         const unsigned char* __restrict__ mask,
                         const float* __restrict__ w0,
                         const float* __restrict__ w1,
                         u16* __restrict__ out, int Dd) {
    const int row = blockIdx.x;
    const float* xr = x + (size_t)row * Dd;
    float ss = 0.f;
    for (int c = threadIdx.x; c < Dd; c += blockDim.x) {
        float vv = xr[c]; ss += vv * vv;
    }
    #pragma unroll
    for (int off = 16; off > 0; off >>= 1) ss += __shfl_xor(ss, off, 32);
    __shared__ float red[8];
    const int wave = threadIdx.x >> 5;
    if ((threadIdx.x & 31) == 0) red[wave] = ss;
    __syncthreads();
    float tot = 0.f;
    #pragma unroll
    for (int i = 0; i < 8; i++) tot += red[i];
    const float scale = rsqrtf(tot / (float)Dd + 1e-5f);
    const float* w = (mask[row] != 0) ? w1 : w0;
    u16* orow = out + (size_t)row * Dd;
    for (int c = threadIdx.x; c < Dd; c += blockDim.x)
        orow[c] = f2bf(xr[c] * scale * w[c]);
}

// ---------------------------------------------------------------------------
// Tiled bf16 GEMM: C[M,N] = A[M,K] x B[K,N], row-major bf16, f32 accum.
// 256 thr = 8 waves; tile 128x128x32; wave = 2x4 subtiles of 16x16.
// Double-buffered LDS; A tile staged by the Tensor Data Mover (async DMA),
// B tile staged transposed (N-major) by the waves.
// Epilogues: 0 bf16, 1 SiLU->bf16, 2 +res->f32, 3 mask-predicated +res->f32
// ---------------------------------------------------------------------------
#define BM 128
#define BN 128
#define BKK 32
#define ATILE (BM * BKK)            // u16 elements per A buffer (4096)
#define BTILE (BN * BKK)

template <int EPI>
__global__ __launch_bounds__(256)
void gemm_bf16_kernel(const u16* __restrict__ A, const u16* __restrict__ Bm,
                      int M, int N, int K,
                      float* __restrict__ Cf, u16* __restrict__ Cb,
                      const float* __restrict__ res,
                      const unsigned char* __restrict__ mask, int want) {
    // single shared object so LDS byte offsets for the TDM are well-defined
    __shared__ __align__(16) u16 smem[2 * ATILE + 2 * BTILE];

    const int tid   = threadIdx.x;
    const int lane  = tid & 31;
    const int wave  = tid >> 5;
    const int half  = lane >> 4;
    const int lx    = lane & 15;
    const int waveM = wave >> 1;          // 0..3 : 32 rows each
    const int waveN = wave & 1;           // 0..1 : 64 cols each
    const int tileM = blockIdx.y * BM;
    const int tileN = blockIdx.x * BN;

    v8f acc[2][4];
    #pragma unroll
    for (int i = 0; i < 2; i++)
        #pragma unroll
        for (int j = 0; j < 4; j++) acc[i][j] = v8zero();

    // --- staging helpers ---
    auto stageA_manual = [&](int k0, u16* Ab) {
        #pragma unroll
        for (int p = 0; p < 2; p++) {
            int idx = (p * 256 + tid) * 8;
            int r = idx >> 5, c = idx & 31;
            *(ushort8*)&Ab[r * BKK + c] =
                *(const ushort8*)(A + (size_t)(tileM + r) * K + k0 + c);
        }
    };
    auto issueA = [&](int k0, int buf) {
#if HAVE_TDM
        if (wave == 0) {
            uint64_t ga = (uint64_t)(uintptr_t)(A + (size_t)tileM * K + k0);
            tdm_load_2d(ga, (uint32_t)(buf * ATILE * 2), BKK, BM, K, M, K);
        }
#else
        stageA_manual(k0, smem + buf * ATILE);
#endif
    };
    auto waitA = [&]() {
#if HAVE_TDM
        if (wave == 0) __builtin_amdgcn_s_wait_tensorcnt(0);
#endif
    };
    auto stageB = [&](int k0, u16* Btb) {    // transposed into Bt[n][k]
        #pragma unroll
        for (int p = 0; p < 2; p++) {
            int idx = (p * 256 + tid) * 8;
            int kr = idx >> 7, nc = idx & 127;
            ushort8 bv =
                *(const ushort8*)(Bm + (size_t)(k0 + kr) * N + tileN + nc);
            #pragma unroll
            for (int j = 0; j < 8; j++) Btb[(nc + j) * BKK + kr] = bv[j];
        }
    };

    // --- prologue: fill buffer 0 ---
    issueA(0, 0);
    stageB(0, smem + 2 * ATILE);
    waitA();
    __syncthreads();

    const int nk = K / BKK;
    for (int it = 0; it < nk; it++) {
        const int buf = it & 1;
        const u16* Ab  = smem + buf * ATILE;
        const u16* Btb = smem + 2 * ATILE + buf * BTILE;

        // kick next tile into the other buffer while we compute
        if (it + 1 < nk) {
            issueA((it + 1) * BKK, buf ^ 1);
            stageB((it + 1) * BKK, smem + 2 * ATILE + (buf ^ 1) * BTILE);
            __builtin_prefetch(A + (size_t)(tileM + (tid >> 1)) * K + (it + 2) * BKK, 0, 1);
        }

        v16bf afr[2], bfr[4];
        #pragma unroll
        for (int sm = 0; sm < 2; sm++) {
            const u16* p = Ab + (waveM * 32 + sm * 16 + lx) * BKK;
            afr[sm] = make_frag(p + half * 8, p + 16 + half * 8);
        }
        #pragma unroll
        for (int sn = 0; sn < 4; sn++) {
            const u16* p = Btb + (waveN * 64 + sn * 16 + lx) * BKK;
            bfr[sn] = make_frag(p + half * 8, p + 16 + half * 8);
        }
        #pragma unroll
        for (int sm = 0; sm < 2; sm++)
            #pragma unroll
            for (int sn = 0; sn < 4; sn++)
                acc[sm][sn] = __builtin_amdgcn_wmma_f32_16x16x32_bf16(
                    false, afr[sm], false, bfr[sn], (short)0, acc[sm][sn],
                    false, false);

        if (it + 1 < nk) waitA();
        __syncthreads();
    }

    // --- epilogue ---
    #pragma unroll
    for (int sm = 0; sm < 2; sm++) {
        #pragma unroll
        for (int sn = 0; sn < 4; sn++) {
            const int cc = tileN + waveN * 64 + sn * 16 + lx;
            #pragma unroll
            for (int vv = 0; vv < 8; vv++) {
                const int rr = tileM + waveM * 32 + sm * 16 + half * 8 + vv;
                const float x = acc[sm][sn][vv];
                const size_t o = (size_t)rr * N + cc;
                if (EPI == 0) {
                    Cb[o] = f2bf(x);
                } else if (EPI == 1) {
                    Cb[o] = f2bf(x / (1.f + __expf(-x)));    // SiLU
                } else if (EPI == 2) {
                    Cf[o] = x + res[o];
                } else {
                    if ((mask[rr] != 0) == (want != 0)) Cf[o] = x + res[o];
                }
            }
        }
    }
}

// ---------------------------------------------------------------------------
// Flash attention, causal, head_dim = 128, bf16 in/out, f32 online softmax.
// Block = 128 thr (4 waves); 64-query tile per block, 64-key tiles streamed.
// ---------------------------------------------------------------------------
#define AQ 64
#define AKV 64

__global__ __launch_bounds__(128)
void flash_attn_kernel(const u16* __restrict__ q, const u16* __restrict__ k,
                       const u16* __restrict__ v, u16* __restrict__ o,
                       int S, int D, int dh) {
    __shared__ __align__(16) u16 Qs[AQ][128];
    __shared__ __align__(16) u16 Ks[AKV][128];
    __shared__ __align__(16) u16 Vt[128][AKV];      // V transposed (d, key)
    __shared__ __align__(16) u16 Ps[4][16][AKV];    // per-wave P scratch

    const int tid  = threadIdx.x;
    const int lane = tid & 31;
    const int wave = tid >> 5;
    const int half = lane >> 4;
    const int lx   = lane & 15;
    const int qt = blockIdx.x;
    const int h  = blockIdx.y;
    const int b  = blockIdx.z;
    const size_t rowbase = (size_t)b * S;
    const int colbase = h * dh;

    // --- stage Q tile ---
    #pragma unroll
    for (int p = 0; p < 8; p++) {
        int idx = (p * 128 + tid) * 8;
        int r = idx >> 7, c = idx & 127;
        *(ushort8*)&Qs[r][c] =
            *(const ushort8*)(q + (rowbase + qt * AQ + r) * D + colbase + c);
    }

    float mrow[8], lrow[8];
    v8f oacc[8];
    #pragma unroll
    for (int i = 0; i < 8; i++) { mrow[i] = -3.0e38f; lrow[i] = 0.f; oacc[i] = v8zero(); }

    const int qb = wave * 16;
    const float sc = 0.08838834764831845f;   // 1/sqrt(128)

    for (int j = 0; j <= qt; j++) {          // causal: only key tiles <= query tile
        __syncthreads();
        // --- stage K (row-major) and V (transposed) ---
        #pragma unroll
        for (int p = 0; p < 8; p++) {
            int idx = (p * 128 + tid) * 8;
            int r = idx >> 7, c = idx & 127;
            *(ushort8*)&Ks[r][c] =
                *(const ushort8*)(k + (rowbase + j * AKV + r) * D + colbase + c);
            ushort8 vvec =
                *(const ushort8*)(v + (rowbase + j * AKV + r) * D + colbase + c);
            #pragma unroll
            for (int e = 0; e < 8; e++) Vt[c + e][r] = vvec[e];
        }
        __syncthreads();
        if (j < qt) {                        // prefetch next K/V tile
            __builtin_prefetch(k + (rowbase + (j + 1) * AKV + (tid >> 1)) * D + colbase, 0, 1);
            __builtin_prefetch(v + (rowbase + (j + 1) * AKV + (tid >> 1)) * D + colbase, 0, 1);
        }

        // --- scores S = Q . K^T (4 key subtiles, contraction d=128) ---
        v8f sacc[4];
        #pragma unroll
        for (int sn = 0; sn < 4; sn++) {
            v8f a = v8zero();
            #pragma unroll
            for (int kc = 0; kc < 4; kc++) {
                v16bf qa = make_frag(&Qs[qb + lx][kc * 32 + half * 8],
                                     &Qs[qb + lx][kc * 32 + 16 + half * 8]);
                v16bf kb = make_frag(&Ks[sn * 16 + lx][kc * 32 + half * 8],
                                     &Ks[sn * 16 + lx][kc * 32 + 16 + half * 8]);
                a = __builtin_amdgcn_wmma_f32_16x16x32_bf16(
                    false, qa, false, kb, (short)0, a, false, false);
            }
            sacc[sn] = a;
        }

        // --- causal mask + online softmax (row reductions across 16 lanes) ---
        #pragma unroll
        for (int vv = 0; vv < 8; vv++) {
            const int qrow = qt * AQ + qb + half * 8 + vv;
            float sv[4];
            float mx = -3.0e38f;
            #pragma unroll
            for (int sn = 0; sn < 4; sn++) {
                int key = j * AKV + sn * 16 + lx;
                float x = sacc[sn][vv] * sc;
                if (key > qrow) x = -3.0e38f;
                sv[sn] = x;
                mx = fmaxf(mx, x);
            }
            #pragma unroll
            for (int off = 1; off < 16; off <<= 1)
                mx = fmaxf(mx, __shfl_xor(mx, off, 32));
            const float mnew  = fmaxf(mrow[vv], mx);
            const float alpha = __expf(mrow[vv] - mnew);
            float psum = 0.f;
            #pragma unroll
            for (int sn = 0; sn < 4; sn++) {
                float p = __expf(sv[sn] - mnew);
                psum += p;
                Ps[wave][half * 8 + vv][sn * 16 + lx] = f2bf(p);
            }
            #pragma unroll
            for (int off = 1; off < 16; off <<= 1)
                psum += __shfl_xor(psum, off, 32);
            lrow[vv] = lrow[vv] * alpha + psum;
            mrow[vv] = mnew;
            #pragma unroll
            for (int dsb = 0; dsb < 8; dsb++) oacc[dsb][vv] *= alpha;
        }

        // Wave-local P round-trip through LDS: DS pipe is in-order per wave.
        asm volatile("s_wait_dscnt 0" ::: "memory");

        // --- O += P . V (key contraction 64 = 2 x 32) ---
        #pragma unroll
        for (int dsb = 0; dsb < 8; dsb++) {
            #pragma unroll
            for (int kc = 0; kc < 2; kc++) {
                v16bf pa = make_frag(&Ps[wave][lx][kc * 32 + half * 8],
                                     &Ps[wave][lx][kc * 32 + 16 + half * 8]);
                v16bf vb = make_frag(&Vt[dsb * 16 + lx][kc * 32 + half * 8],
                                     &Vt[dsb * 16 + lx][kc * 32 + 16 + half * 8]);
                oacc[dsb] = __builtin_amdgcn_wmma_f32_16x16x32_bf16(
                    false, pa, false, vb, (short)0, oacc[dsb], false, false);
            }
        }
    }

    // --- epilogue: O /= l, store bf16 ---
    #pragma unroll
    for (int dsb = 0; dsb < 8; dsb++) {
        #pragma unroll
        for (int vv = 0; vv < 8; vv++) {
            const int qrow = qt * AQ + qb + half * 8 + vv;
            const float val = oacc[dsb][vv] / lrow[vv];
            o[(rowbase + qrow) * D + colbase + dsb * 16 + lx] = f2bf(val);
        }
    }
}

// ---------------------------------------------------------------------------
// Host-side orchestration
// ---------------------------------------------------------------------------
extern "C" void kernel_launch(void* const* d_in, const int* in_sizes, int n_in,
                              void* d_out, int out_size, void* d_ws, size_t ws_size,
                              hipStream_t stream) {
    (void)in_sizes; (void)n_in; (void)out_size; (void)ws_size;
    const int Bb = 2, S = 2048, Dd = 2048, F = 8192, H = 16, dh = 128;
    const int BS = Bb * S;

    const float* hidden       = (const float*)d_in[0];
    const unsigned char* mask = (const unsigned char*)d_in[1];  // jax bool = 1 byte
    const float* w_ln_in      = (const float*)d_in[2];
    const float* w_ln_in_a    = (const float*)d_in[3];
    const float* w_ln_post    = (const float*)d_in[4];
    const float* w_ln_post_a  = (const float*)d_in[5];
    const float* Wq = (const float*)d_in[6];
    const float* Wk = (const float*)d_in[7];
    const float* Wv = (const float*)d_in[8];
    const float* Wo = (const float*)d_in[9];
    const float* Wfc  = (const float*)d_in[10];
    const float* Wpr  = (const float*)d_in[11];
    const float* Wfca = (const float*)d_in[12];
    const float* Wpra = (const float*)d_in[13];
    float* out = (float*)d_out;

    // workspace carve-out (256B aligned)
    uint8_t* ws = (uint8_t*)d_ws;
    size_t off = 0;
    auto carve = [&](size_t bytes) -> void* {
        void* p = ws + off;
        off = (off + bytes + 255) & ~(size_t)255;
        return p;
    };
    const size_t szDD = (size_t)Dd * Dd * 2;
    const size_t szDF = (size_t)Dd * F * 2;
    u16* wqb  = (u16*)carve(szDD);
    u16* wkb  = (u16*)carve(szDD);
    u16* wvb  = (u16*)carve(szDD);
    u16* wob  = (u16*)carve(szDD);
    u16* wfcb  = (u16*)carve(szDF);
    u16* wfcab = (u16*)carve(szDF);
    u16* wprb  = (u16*)carve(szDF);
    u16* wprab = (u16*)carve(szDF);
    u16* x0  = (u16*)carve((size_t)BS * Dd * 2);
    u16* qb  = (u16*)carve((size_t)BS * Dd * 2);
    u16* kb  = (u16*)carve((size_t)BS * Dd * 2);
    u16* vb  = (u16*)carve((size_t)BS * Dd * 2);
    u16* ao  = (u16*)carve((size_t)BS * Dd * 2);
    float* hbuf = (float*)carve((size_t)BS * Dd * 4);
    u16* xt  = (u16*)carve((size_t)BS * Dd * 2);
    u16* xa  = (u16*)carve((size_t)BS * Dd * 2);
    u16* t1  = (u16*)carve((size_t)BS * F * 2);
    u16* t2  = (u16*)carve((size_t)BS * F * 2);

    // 1) convert weights to bf16
    const float* srcs[8] = {Wq, Wk, Wv, Wo, Wfc, Wfca, Wpr, Wpra};
    u16* dsts[8]         = {wqb, wkb, wvb, wob, wfcb, wfcab, wprb, wprab};
    const long  ns[8]    = {(long)Dd*Dd, (long)Dd*Dd, (long)Dd*Dd, (long)Dd*Dd,
                            (long)Dd*F, (long)Dd*F, (long)F*Dd, (long)F*Dd};
    for (int i = 0; i < 8; i++) {
        long n = ns[i];
        int grid = (int)((n + 255) / 256);
        if (grid > 4096) grid = 4096;
        cvt_f32_bf16_kernel<<<grid, 256, 0, stream>>>(srcs[i], dsts[i], n);
    }

    // 2) dual input RMSNorm
    rmsnorm_dual_kernel<<<BS, 256, 0, stream>>>(hidden, mask, w_ln_in, w_ln_in_a, x0, Dd);

    // 3) Q/K/V projections (bf16 out)
    dim3 gDD(Dd / BN, BS / BM);
    gemm_bf16_kernel<0><<<gDD, 256, 0, stream>>>(x0, wqb, BS, Dd, Dd, nullptr, qb, nullptr, nullptr, 0);
    gemm_bf16_kernel<0><<<gDD, 256, 0, stream>>>(x0, wkb, BS, Dd, Dd, nullptr, kb, nullptr, nullptr, 0);
    gemm_bf16_kernel<0><<<gDD, 256, 0, stream>>>(x0, wvb, BS, Dd, Dd, nullptr, vb, nullptr, nullptr, 0);

    // 4) causal flash attention
    dim3 gAtt(S / AQ, H, Bb);
    flash_attn_kernel<<<gAtt, 128, 0, stream>>>(qb, kb, vb, ao, S, Dd, dh);

    // 5) output projection + residual -> h (f32)
    gemm_bf16_kernel<2><<<gDD, 256, 0, stream>>>(ao, wob, BS, Dd, Dd, hbuf, nullptr, hidden, nullptr, 0);

    // 6) post RMSNorms (text / audio weight sets, all rows)
    rmsnorm_dual_kernel<<<BS, 256, 0, stream>>>(hbuf, mask, w_ln_post,   w_ln_post,   xt, Dd);
    rmsnorm_dual_kernel<<<BS, 256, 0, stream>>>(hbuf, mask, w_ln_post_a, w_ln_post_a, xa, Dd);

    // 7) fc GEMMs with fused SiLU -> bf16
    dim3 gDF(F / BN, BS / BM);
    gemm_bf16_kernel<1><<<gDF, 256, 0, stream>>>(xt, wfcb,  BS, F, Dd, nullptr, t1, nullptr, nullptr, 0);
    gemm_bf16_kernel<1><<<gDF, 256, 0, stream>>>(xa, wfcab, BS, F, Dd, nullptr, t2, nullptr, nullptr, 0);

    // 8) proj GEMMs, mask-predicated residual write: together they cover every row
    dim3 gFD(Dd / BN, BS / BM);
    gemm_bf16_kernel<3><<<gFD, 256, 0, stream>>>(t1, wprb,  BS, Dd, F, out, nullptr, hbuf, mask, 0);
    gemm_bf16_kernel<3><<<gFD, 256, 0, stream>>>(t2, wprab, BS, Dd, F, out, nullptr, hbuf, mask, 1);
}
